// ProDAAE_26319559590754
// MI455X (gfx1250) — matrix operand order repaired and checked by reading, not compiled, not run on hardware
//
#include <hip/hip_runtime.h>
#include <hip/hip_bf16.h>

// ---------- types ----------
typedef __attribute__((ext_vector_type(16))) __bf16        bf16x16;
typedef __attribute__((ext_vector_type(8)))  float         f32x8;
typedef __attribute__((ext_vector_type(4)))  unsigned int  u32x4;

union FragU {
    bf16x16        v;
    unsigned short u[16];
    u32x4          q[2];
};

__device__ __forceinline__ unsigned short f2bf(float f) {
    unsigned int u = __float_as_uint(f);
    unsigned int r = u + 0x7FFFu + ((u >> 16) & 1u);   // RNE
    return (unsigned short)(r >> 16);
}

#define WMMA_BF16(ACC, AF, BF) \
    ACC = __builtin_amdgcn_wmma_f32_16x16x32_bf16(false, (AF).v, false, (BF).v, \
                                                  (short)0, ACC, false, false)

__device__ __forceinline__ void load_a(FragU& f, const unsigned short* p) {
    f.q[0] = *(const u32x4*)(p);        // K = half*16 .. +7
    f.q[1] = *(const u32x4*)(p + 8);    // K = half*16 + 8 .. +15
}
__device__ __forceinline__ void load_b(FragU& f, const unsigned short* p) {
    f.q[0] = *(const u32x4*)(p);        // K = half*16 .. +7
    f.q[1] = *(const u32x4*)(p + 8);    // K = half*16 + 8 .. +15
}

// ---------- f32 -> bf16 convert (row-major, A-role: adjacency) ----------
__global__ void k_cvt(const float* __restrict__ src, unsigned short* __restrict__ dst,
                      long long total) {
    long long i = (long long)blockIdx.x * blockDim.x + threadIdx.x;
    if (i >= total) return;
    dst[i] = f2bf(src[i]);
}

// ---------- f32 [K,cols] -> bf16 transposed [dcols,K] (B-role weights) ----------
__global__ void k_cvt_t(const float* __restrict__ src, unsigned short* __restrict__ dst,
                        int K, int cols, int dcols) {
    long long i = (long long)blockIdx.x * blockDim.x + threadIdx.x;
    long long total = (long long)dcols * K;
    if (i >= total) return;
    int k = (int)(i % K);
    int n = (int)(i / K);
    dst[i] = (n < cols) ? f2bf(src[(long long)k * cols + n]) : (unsigned short)0;
}

// ---------- layer-0 projection: G0^T = (x @ W0)^T, per batch [64,512] ----------
__global__ void k_xw0_t(const float* __restrict__ x, const float* __restrict__ W0,
                        unsigned short* __restrict__ g, int total) {
    int i = blockIdx.x * blockDim.x + threadIdx.x;   // i = ((b*64)+c)*512 + r
    if (i >= total) return;
    int r = i & 511;
    int c = (i >> 9) & 63;
    int b = i >> 15;
    int rg = b * 512 + r;
    float v = x[2 * rg] * W0[c] + x[2 * rg + 1] * W0[64 + c];
    g[i] = f2bf(v);
}

// ---------- generic WMMA bf16 GEMM, wave = one 32x32 D tile ----------
// A: [batches][M,K] row-major bf16.  B: [batches][N,K] TRANSPOSED bf16.
// Requires: M%32==0, N%32==0, K%64==0.
struct GemmP {
    const unsigned short* A;
    const unsigned short* B;      // transposed [N,K]
    const float* bias;            // [N] or null
    const float* mask;            // per-row mask or null
    void* C;
    int M, N, K;
    int lda, ldb, ldc;            // ldb = K stride of B^T; TRANS: ldc = M stride of C^T
    long long sA, sB, sC;
    int batches;
    int maskStride;
    int dvalid;                   // valid output columns (predicated store)
    int flags;                    // 1=relu 2=mask 4=bf16-out 8=store-transposed(packed)
};

__global__ void k_gemm(GemmP p) {
    const int lane = threadIdx.x & 31;
    const int wave = (blockIdx.x << 3) + (threadIdx.x >> 5);
    const int tilesM = p.M >> 5, tilesN = p.N >> 5;      // 32x32 tiles
    const int perB = tilesM * tilesN;
    if (wave >= perB * p.batches) return;
    const int b  = wave / perB;
    const int t  = wave - b * perB;
    const int tn = t % tilesN;
    const int tm = t / tilesN;
    const int half = lane >> 4, l16 = lane & 15;

    // A frag rows: lanes 0-15 -> K [0..15]; lanes 16-31 -> [16..31]
    const unsigned short* __restrict__ A0 =
        p.A + (long long)b * p.sA + (long long)(tm * 32 + l16) * p.lda + half * 16;
    const unsigned short* __restrict__ A1 = A0 + 16ll * p.lda;
    // B frag (B^T): lane column n, contiguous K chunk at half*16
    const unsigned short* __restrict__ B0 =
        p.B + (long long)b * p.sB + (long long)(tn * 32 + l16) * p.ldb + half * 16;
    const unsigned short* __restrict__ B1 = B0 + 16ll * p.ldb;

    f32x8 c00 = {0.f,0.f,0.f,0.f,0.f,0.f,0.f,0.f};
    f32x8 c01 = c00, c10 = c00, c11 = c00;

    // explicit double buffering: K%64==0
    FragU aA0, aA1, bA0, bA1;      // even 32-step
    FragU aB0, aB1, bB0, bB1;      // odd 32-step
    load_a(aA0, A0); load_a(aA1, A1);
    load_b(bA0, B0); load_b(bA1, B1);
    for (int kk = 0; kk < p.K; kk += 64) {
        // issue odd-step loads while even-step WMMAs run
        load_a(aB0, A0 + kk + 32); load_a(aB1, A1 + kk + 32);
        load_b(bB0, B0 + kk + 32); load_b(bB1, B1 + kk + 32);
        __builtin_prefetch(A0 + kk + 64, 0, 1);
        __builtin_prefetch(B0 + kk + 64, 0, 1);
        WMMA_BF16(c00, aA0, bA0);
        WMMA_BF16(c01, aA0, bA1);
        WMMA_BF16(c10, aA1, bA0);
        WMMA_BF16(c11, aA1, bA1);
        // issue next even-step loads while odd-step WMMAs run
        if (kk + 64 < p.K) {
            load_a(aA0, A0 + kk + 64); load_a(aA1, A1 + kk + 64);
            load_b(bA0, B0 + kk + 64); load_b(bA1, B1 + kk + 64);
        }
        WMMA_BF16(c00, aB0, bB0);
        WMMA_BF16(c01, aB0, bB1);
        WMMA_BF16(c10, aB1, bB0);
        WMMA_BF16(c11, aB1, bB1);
    }

    f32x8 accs[2][2] = {{c00, c01}, {c10, c11}};
#pragma unroll
    for (int mi = 0; mi < 2; ++mi) {
#pragma unroll
        for (int ni = 0; ni < 2; ++ni) {
            const int n = tn * 32 + ni * 16 + l16;
            const bool nv = n < p.dvalid;
            const float bias = (p.bias && nv) ? p.bias[n] : 0.0f;
            if (p.flags & 8) {
                // packed transposed bf16 store: 8 consecutive rows per lane
                union { unsigned short s[8]; u32x4 q; } o;
#pragma unroll
                for (int v = 0; v < 8; ++v) {
                    float c = accs[mi][ni][v] + bias;
                    if (p.flags & 1) c = fmaxf(c, 0.0f);
                    o.s[v] = f2bf(c);
                }
                long long off = (long long)b * p.sC + (long long)n * p.ldc
                              + tm * 32 + mi * 16 + (half << 3);
                *(u32x4*)((unsigned short*)p.C + off) = o.q;
            } else {
#pragma unroll
                for (int v = 0; v < 8; ++v) {
                    int m = tm * 32 + mi * 16 + (half << 3) + v;
                    float c = accs[mi][ni][v] + bias;
                    if (p.flags & 1) c = fmaxf(c, 0.0f);
                    if (p.flags & 2) c *= p.mask[(long long)b * p.maskStride + m];
                    if (nv) {
                        long long off = (long long)b * p.sC + (long long)m * p.ldc + n;
                        if (p.flags & 4) ((unsigned short*)p.C)[off] = f2bf(c);
                        else             ((float*)p.C)[off] = c;
                    }
                }
            }
        }
    }
}

// ---------- fused mu/lv GEMM + reparametrization: z = mu + eps*exp(0.5*lv) ----------
__global__ void k_zgemm(const unsigned short* __restrict__ A,     // [M,256] bf16
                        const unsigned short* __restrict__ BmuT,  // [128,256] bf16 (W^T)
                        const unsigned short* __restrict__ BlvT,  // [128,256] bf16 (W^T)
                        const float* __restrict__ mub, const float* __restrict__ lvb,
                        const float* __restrict__ eps,            // [M,128] f32
                        unsigned short* __restrict__ Z, int M) {
    const int lane = threadIdx.x & 31;
    const int wave = (blockIdx.x << 3) + (threadIdx.x >> 5);
    const int tilesN = 8;                         // 128/16
    const int tilesM = M >> 4;
    if (wave >= tilesM * tilesN) return;
    const int tn = wave % tilesN, tm = wave / tilesN;
    const int half = lane >> 4, l16 = lane & 15;

    const unsigned short* a   = A    + (long long)(tm * 16 + l16) * 256 + half * 16;
    const unsigned short* bmu = BmuT + (long long)(tn * 16 + l16) * 256 + half * 16;
    const unsigned short* blv = BlvT + (long long)(tn * 16 + l16) * 256 + half * 16;

    f32x8 accMu = {0.f,0.f,0.f,0.f,0.f,0.f,0.f,0.f};
    f32x8 accLv = accMu;
    FragU afA, bmA, blA, afB, bmB, blB;
    load_a(afA, a); load_b(bmA, bmu); load_b(blA, blv);
    for (int kk = 0; kk < 256; kk += 64) {
        load_a(afB, a + kk + 32); load_b(bmB, bmu + kk + 32); load_b(blB, blv + kk + 32);
        WMMA_BF16(accMu, afA, bmA);
        WMMA_BF16(accLv, afA, blA);
        if (kk + 64 < 256) {
            load_a(afA, a + kk + 64); load_b(bmA, bmu + kk + 64); load_b(blA, blv + kk + 64);
        }
        WMMA_BF16(accMu, afB, bmB);
        WMMA_BF16(accLv, afB, blB);
    }
    const int n = tn * 16 + l16;
#pragma unroll
    for (int v = 0; v < 8; ++v) {
        int m = tm * 16 + (half << 3) + v;
        float mu = accMu[v] + mub[n];
        float lv = accLv[v] + lvb[n];
        float z  = mu + eps[(long long)m * 128 + n] * __expf(0.5f * lv);
        Z[(long long)m * 128 + n] = f2bf(z);
    }
}

// ---------- host side ----------
static void launch_gemm(const unsigned short* A, int lda, long long sA,
                        const unsigned short* B, int ldb, long long sB,
                        const float* bias, const float* mask, int maskStride,
                        void* C, int ldc, long long sC,
                        int M, int N, int K, int batches, int dvalid, int flags,
                        hipStream_t stream) {
    GemmP p;
    p.A = A; p.B = B; p.bias = bias; p.mask = mask; p.C = C;
    p.M = M; p.N = N; p.K = K;
    p.lda = lda; p.ldb = ldb; p.ldc = ldc;
    p.sA = sA; p.sB = sB; p.sC = sC;
    p.batches = batches; p.maskStride = maskStride;
    p.dvalid = dvalid; p.flags = flags;
    int waves = batches * (M >> 5) * (N >> 5);
    int blocks = (waves + 7) >> 3;
    k_gemm<<<blocks, 256, 0, stream>>>(p);
}

extern "C" void kernel_launch(void* const* d_in, const int* in_sizes, int n_in,
                              void* d_out, int out_size, void* d_ws, size_t ws_size,
                              hipStream_t stream) {
    (void)in_sizes; (void)n_in; (void)out_size;
    const int Bn = 64, Nn = 512, Mrows = Bn * Nn;   // 32768
    if (ws_size < (size_t)70 * 1024 * 1024) return; // need ~66MB scratch

    const float* x      = (const float*)d_in[0];
    const float* w_adj  = (const float*)d_in[1];
    const float* mask   = (const float*)d_in[2];
    const float* eps    = (const float*)d_in[3];
    const float* enc_W0 = (const float*)d_in[4];
    const float* enc_b0 = (const float*)d_in[5];
    const float* enc_W1 = (const float*)d_in[6];
    const float* enc_b1 = (const float*)d_in[7];
    const float* enc_W2 = (const float*)d_in[8];
    const float* enc_b2 = (const float*)d_in[9];
    const float* mu_W   = (const float*)d_in[10];
    const float* mu_b   = (const float*)d_in[11];
    const float* lv_W   = (const float*)d_in[12];
    const float* lv_b   = (const float*)d_in[13];

    char* ws = (char*)d_ws;
    unsigned short* adjb = (unsigned short*)ws;                       // 32MB
    unsigned short* buf1 = (unsigned short*)(ws + (32ll << 20));      // 16MB
    unsigned short* buf2 = (unsigned short*)(ws + (48ll << 20));      // 16MB
    unsigned short* wtb  = (unsigned short*)(ws + (64ll << 20));      // ~1.1MB

    // adjacency -> bf16 (A operand; reused by all 3 encoder layers, L2-resident)
    {
        long long tot = (long long)Bn * Nn * Nn;
        k_cvt<<<(int)((tot + 255) / 256), 256, 0, stream>>>(w_adj, adjb, tot);
    }

    auto convT = [&](const float* src, unsigned short* dst, int K, int cols, int dcols) {
        long long tot = (long long)dcols * K;
        k_cvt_t<<<(int)((tot + 255) / 256), 256, 0, stream>>>(src, dst, K, cols, dcols);
    };

    // converted transposed (and padded) weights: layout [Npad, K]
    long long o = 0;
    unsigned short* w1bT  = wtb + o; o += 128 * 64;
    unsigned short* w2bT  = wtb + o; o += 256 * 128;
    unsigned short* muWbT = wtb + o; o += 128 * 256;
    unsigned short* lvWbT = wtb + o; o += 128 * 256;
    unsigned short* hW0T[3]; unsigned short* hW1T[3]; unsigned short* hW2T[3];
    const int hPad[3] = {32, 32, 512};     // aa:21->32, ss:8->32, wa:512 (N%32==0)
    const int hOut[3] = {21, 8, 512};
    for (int h = 0; h < 3; ++h) {
        hW0T[h] = wtb + o; o += 256 * 128;
        hW1T[h] = wtb + o; o += 256 * 256;
        hW2T[h] = wtb + o; o += (long long)hPad[h] * 256;
    }
    convT(enc_W1, w1bT, 64, 128, 128);
    convT(enc_W2, w2bT, 128, 256, 256);
    convT(mu_W, muWbT, 256, 128, 128);
    convT(lv_W, lvWbT, 256, 128, 128);
    for (int h = 0; h < 3; ++h) {
        convT((const float*)d_in[14 + 6 * h + 0], hW0T[h], 128, 256, 256);
        convT((const float*)d_in[14 + 6 * h + 2], hW1T[h], 256, 256, 256);
        convT((const float*)d_in[14 + 6 * h + 4], hW2T[h], 256, hOut[h], hPad[h]);
    }

    const int RELU = 1, MASK = 2, OBF16 = 4, TRANS = 8;

    // ---- encoder: H = relu(adj@(h@W) + b)*m ----
    // G0^T = (x@W0)^T -> buf1 [64][64,512]
    k_xw0_t<<<(Mrows * 64) / 256, 256, 0, stream>>>(x, enc_W0, buf1, Mrows * 64);
    // H0 = relu(adj@G0+b0)*m -> buf2 row-major [64][512,64]
    launch_gemm(adjb, 512, 512ll * 512, buf1, 512, 64ll * 512, enc_b0, mask, 512,
                buf2, 64, 512ll * 64, 512, 64, 512, Bn, 64, RELU | MASK | OBF16, stream);
    // G1^T = (H0@W1)^T -> buf1 [64][128,512]  (batched, shared B)
    launch_gemm(buf2, 64, 512ll * 64, w1bT, 64, 0, nullptr, nullptr, 0,
                buf1, 512, 128ll * 512, 512, 128, 64, Bn, 128, OBF16 | TRANS, stream);
    // H1 -> buf2 [64][512,128]
    launch_gemm(adjb, 512, 512ll * 512, buf1, 512, 128ll * 512, enc_b1, mask, 512,
                buf2, 128, 512ll * 128, 512, 128, 512, Bn, 128, RELU | MASK | OBF16, stream);
    // G2^T = (H1@W2)^T -> buf1 [64][256,512]
    launch_gemm(buf2, 128, 512ll * 128, w2bT, 128, 0, nullptr, nullptr, 0,
                buf1, 512, 256ll * 512, 512, 256, 128, Bn, 256, OBF16 | TRANS, stream);
    // H2 -> buf2 [64][512,256]
    launch_gemm(adjb, 512, 512ll * 512, buf1, 512, 256ll * 512, enc_b2, mask, 512,
                buf2, 256, 512ll * 256, 512, 256, 512, Bn, 256, RELU | MASK | OBF16, stream);

    // ---- z = mu + eps*exp(0.5*lv) -> buf1 [32768,128] bf16 ----
    {
        int waves = (Mrows >> 4) * 8;
        k_zgemm<<<(waves + 7) >> 3, 256, 0, stream>>>(buf2, muWbT, lvWbT, mu_b, lv_b,
                                                      eps, buf1, Mrows);
    }

    // ---- decoder heads into d_out [32768, 541] ----
    unsigned short* T1 = adjb;   // adjacency no longer needed; reuse 16MB
    int colofs = 0;
    for (int h = 0; h < 3; ++h) {
        const float* b0 = (const float*)d_in[14 + 6 * h + 1];
        const float* b1 = (const float*)d_in[14 + 6 * h + 3];
        const float* b2 = (const float*)d_in[14 + 6 * h + 5];
        // T0 = relu(z@W0+b0) -> buf2 [32768,256]
        launch_gemm(buf1, 128, 0, hW0T[h], 128, 0, b0, nullptr, 0,
                    buf2, 256, 0, Mrows, 256, 128, 1, 256, RELU | OBF16, stream);
        // T1 = relu(T0@W1+b1) -> T1 [32768,256]
        launch_gemm(buf2, 256, 0, hW1T[h], 256, 0, b1, nullptr, 0,
                    T1, 256, 0, Mrows, 256, 256, 1, 256, RELU | OBF16, stream);
        // out = T1@W2+b2 -> d_out (f32, strided into 541-wide concat)
        launch_gemm(T1, 256, 0, hW2T[h], 256, 0, b2, nullptr, 0,
                    (float*)d_out + colofs, 541, 0, Mrows, hPad[h], 256, 1,
                    hOut[h], 0, stream);
        colofs += hOut[h];
    }
}
